// ResidualAdvection_1099511628422
// MI455X (gfx1250) — matrix operand order repaired and checked by reading, not compiled, not run on hardware
//
#include <hip/hip_runtime.h>
#include <hip/hip_bf16.h>

typedef __attribute__((ext_vector_type(16))) _Float16 v16h;
typedef __attribute__((ext_vector_type(8)))  _Float16 v8h;
typedef __attribute__((ext_vector_type(8)))  float    v8f;

#define BATCH 8
#define HH 512
#define WW 512
#define TILE 16            // output tile is 16x16 pixels
#define NTHREADS 256       // 8 wave32
#define OSTRIDE 19         // offsBuf row stride (gcd(19,64)=1 -> conflict free)

// Fully fused: conv1(3->16,3x3)+GELU -> conv2(16->18,3x3) via WMMA -> deformable bilinear sample.
// K ordering for the conv2 GEMM: k = tap*16 + ci  (tap in 0..8, ci in 0..15), K padded 144->160.
// h stored channel-innermost so each A-fragment 8-element group is one aligned b128 LDS load.
__global__ __launch_bounds__(NTHREADS)
void fused_deform_kernel(const float* __restrict__ pm25,
                         const float* __restrict__ wind,
                         const float* __restrict__ topo,
                         const float* __restrict__ w1,   // (16,3,3,3)
                         const float* __restrict__ b1,   // (16)
                         const float* __restrict__ w2,   // (18,16,3,3)
                         const float* __restrict__ b2,   // (18)
                         const float* __restrict__ wgt,  // (1,1,3,3) -> 9
                         float* __restrict__ out)
{
    // ---- LDS ----
    __shared__ float    inBuf[3 * 20 * 20];                        // input tile, halo 2
    __shared__ __align__(16) _Float16 h2[18 * 18 * 16];            // [y][x][ci] f16, halo 1
    __shared__ __align__(16) _Float16 w2hT[32 * 160];              // [n][k] f16, zero padded
    __shared__ float    offsBuf[256 * OSTRIDE];                    // conv2 out, [pixel][channel]
    __shared__ float    w1s[16 * 27];
    __shared__ float    b1s[16];
    __shared__ float    b2s[18];
    __shared__ float    wks[9];

    const int tid    = threadIdx.x;
    const int tileX0 = blockIdx.x * TILE;
    const int tileY0 = blockIdx.y * TILE;
    const int b      = blockIdx.z;

    // ---- Stage 0: weights to LDS ----
    for (int i = tid; i < 16 * 27; i += NTHREADS) w1s[i] = w1[i];
    if (tid < 16) b1s[tid] = b1[tid];
    if (tid < 18) b2s[tid] = b2[tid];
    if (tid < 9)  wks[tid] = wgt[tid];
    // w2 (co,ci,ky,kx) -> w2hT[n = co][k = tap*16 + ci]; zero pad n>=18, k>=144
    for (int i = tid; i < 32 * 160; i += NTHREADS) {
        const int n   = i / 160;
        const int k   = i - n * 160;
        const int tap = k >> 4;
        const int ci  = k & 15;
        w2hT[i] = (n < 18 && tap < 9) ? (_Float16)w2[n * 144 + ci * 9 + tap]
                                      : (_Float16)0.0f;
    }

    // ---- Stage 1: load 3-channel input tile with halo 2, zero-padded ----
    for (int i = tid; i < 3 * 400; i += NTHREADS) {
        const int c   = i / 400;
        const int rem = i - c * 400;
        const int yy  = rem / 20;
        const int xx  = rem - yy * 20;
        const int gy  = tileY0 - 2 + yy;
        const int gx  = tileX0 - 2 + xx;
        float v = 0.0f;
        if (gy >= 0 && gy < HH && gx >= 0 && gx < WW) {
            v = (c < 2) ? wind[((b * 2 + c) * HH + gy) * WW + gx]
                        : topo[(b * HH + gy) * WW + gx];
        }
        inBuf[i] = v;
    }
    __syncthreads();

    // ---- Stage 2: conv1 + exact GELU -> h2 (f16, channel-innermost), 18x18 spatial ----
    for (int i = tid; i < 324 * 16; i += NTHREADS) {
        const int pix = i >> 4;          // 0..323
        const int c2  = i & 15;
        const int yy  = pix / 18;
        const int xx  = pix - yy * 18;
        const int gy  = tileY0 - 1 + yy;
        const int gx  = tileX0 - 1 + xx;
        float v = 0.0f;
        if (gy >= 0 && gy < HH && gx >= 0 && gx < WW) {
            float acc = b1s[c2];
            #pragma unroll
            for (int ci = 0; ci < 3; ++ci)
                #pragma unroll
                for (int ty = 0; ty < 3; ++ty)
                    #pragma unroll
                    for (int tx = 0; tx < 3; ++tx)
                        acc += inBuf[ci * 400 + (yy + ty) * 20 + (xx + tx)] *
                               w1s[c2 * 27 + ci * 9 + ty * 3 + tx];
            v = 0.5f * acc * (1.0f + erff(acc * 0.70710678118654752f)); // exact GELU
        }
        h2[pix * 16 + c2] = (_Float16)v;
    }
    __syncthreads();

    // ---- Stage 3: conv2 as im2col GEMM on WMMA (M=256 px, N=18->2x16, K=160=5x32) ----
    {
        const int wave = tid >> 5;
        const int lane = tid & 31;
        const int half = lane >> 4;      // 0 or 1
        const int l15  = lane & 15;

        for (int mt = wave * 2; mt < wave * 2 + 2; ++mt) {
            const int pA = mt * 16 + l15;        // A-fragment row -> pixel
            const int py = pA >> 4;
            const int px = pA & 15;
            v8f c0 = {};
            v8f c1 = {};
            #pragma unroll
            for (int s = 0; s < 5; ++s) {
                // A fragment: elements 0..7  -> tap 2s,   ci = half*8 + e
                //             elements 8..15 -> tap 2s+1, ci = half*8 + e-8
                const int t0  = 2 * s;
                const int t1  = 2 * s + 1;
                const int ty0 = t0 / 3, tx0 = t0 - ty0 * 3;
                const v8h alo = *(const v8h*)&h2[(((py + ty0) * 18) + (px + tx0)) * 16 + half * 8];
                v8h ahi = {};
                if (t1 < 9) {            // compile-time after unroll (false only for s==4)
                    const int ty1 = t1 / 3, tx1 = t1 - ty1 * 3;
                    ahi = *(const v8h*)&h2[(((py + ty1) * 18) + (px + tx1)) * 16 + half * 8];
                }
                v16h a;
                #pragma unroll
                for (int e = 0; e < 8; ++e) { a[e] = alo[e]; a[8 + e] = ahi[e]; }

                // B fragments: lane column n, k = s*32 + half*8 + 0..7 and s*32 + 16 + half*8 + 0..7
                const int kb = s * 32 + half * 8;
                const v8h b0lo = *(const v8h*)&w2hT[l15 * 160 + kb];
                const v8h b0hi = *(const v8h*)&w2hT[l15 * 160 + kb + 16];
                const v8h b1lo = *(const v8h*)&w2hT[(16 + l15) * 160 + kb];
                const v8h b1hi = *(const v8h*)&w2hT[(16 + l15) * 160 + kb + 16];
                v16h bf0, bf1;
                #pragma unroll
                for (int e = 0; e < 8; ++e) {
                    bf0[e] = b0lo[e]; bf0[8 + e] = b0hi[e];
                    bf1[e] = b1lo[e]; bf1[8 + e] = b1hi[e];
                }

                c0 = __builtin_amdgcn_wmma_f32_16x16x32_f16(false, a, false, bf0,
                                                            (short)0, c0, false, false);
                c1 = __builtin_amdgcn_wmma_f32_16x16x32_f16(false, a, false, bf1,
                                                            (short)0, c1, false, false);
            }
            // D layout: VGPR r -> M = half*8 + r, N = l15
            #pragma unroll
            for (int r = 0; r < 8; ++r) {
                const int pm = mt * 16 + half * 8 + r;
                offsBuf[pm * OSTRIDE + l15] = c0[r] + b2s[l15];
                if (l15 + 16 < 18)
                    offsBuf[pm * OSTRIDE + 16 + l15] = c1[r] + b2s[16 + l15];
            }
        }
    }
    __syncthreads();

    // ---- Stage 4: deformable bilinear sampling of pm25, one pixel per thread ----
    {
        const int p  = tid;            // 0..255
        const int py = p >> 4;
        const int px = p & 15;
        const int gy = tileY0 + py;
        const int gx = tileX0 + px;
        const float* img = pm25 + (size_t)b * HH * WW;

        auto ld = [&](int yi, int xi) -> float {
            return (yi >= 0 && yi < HH && xi >= 0 && xi < WW)
                       ? img[yi * WW + xi] : 0.0f;
        };

        float acc = 0.0f;
        #pragma unroll
        for (int k = 0; k < 9; ++k) {
            const float dy = offsBuf[p * OSTRIDE + 2 * k];
            const float dx = offsBuf[p * OSTRIDE + 2 * k + 1];
            const float pyf = (float)(gy + (k / 3) - 1) + dy;
            const float pxf = (float)(gx + (k % 3) - 1) + dx;
            const float y0f = floorf(pyf);
            const float x0f = floorf(pxf);
            const float wy1 = pyf - y0f;
            const float wx1 = pxf - x0f;
            const int y0 = (int)y0f;
            const int x0 = (int)x0f;
            const float v00 = ld(y0,     x0    );
            const float v01 = ld(y0,     x0 + 1);
            const float v10 = ld(y0 + 1, x0    );
            const float v11 = ld(y0 + 1, x0 + 1);
            const float samp = (1.0f - wy1) * (1.0f - wx1) * v00 +
                               (1.0f - wy1) * wx1          * v01 +
                               wy1          * (1.0f - wx1) * v10 +
                               wy1          * wx1          * v11;
            acc += samp * wks[k];
        }
        out[((size_t)b * HH + gy) * WW + gx] = acc;
    }
}

extern "C" void kernel_launch(void* const* d_in, const int* in_sizes, int n_in,
                              void* d_out, int out_size, void* d_ws, size_t ws_size,
                              hipStream_t stream) {
    (void)in_sizes; (void)n_in; (void)d_ws; (void)ws_size; (void)out_size;
    const float* pm25 = (const float*)d_in[0];
    const float* wind = (const float*)d_in[1];
    const float* topo = (const float*)d_in[2];
    const float* w1   = (const float*)d_in[3];
    const float* b1   = (const float*)d_in[4];
    const float* w2   = (const float*)d_in[5];
    const float* b2   = (const float*)d_in[6];
    const float* wgt  = (const float*)d_in[7];
    float* out = (float*)d_out;

    dim3 grid(WW / TILE, HH / TILE, BATCH);   // 32 x 32 x 8
    dim3 block(NTHREADS);
    fused_deform_kernel<<<grid, block, 0, stream>>>(pm25, wind, topo, w1, b1, w2, b2, wgt, out);
}
